// CrossAttention_5162550690345
// MI455X (gfx1250) — compile-verified, matrix operand order checked by reading, not run on hardware
//
#include <hip/hip_runtime.h>
#include <stdint.h>

#define DEV __device__ __forceinline__

typedef unsigned short u16t;
typedef __bf16 v16bf __attribute__((ext_vector_type(16)));
typedef float v8f __attribute__((ext_vector_type(8)));
typedef unsigned int v8u __attribute__((ext_vector_type(8)));

namespace cfg {
constexpr int B = 32, T = 1024, N = 256, D = 1024, TD = 768, H = 16, C = 64, TE = 2048;
constexpr int D2 = 2 * D;
}

// round-to-nearest-even f32 -> bf16 bits
DEV u16t f2bf(float f) {
  uint32_t u = __float_as_uint(f);
  u += 0x7FFFu + ((u >> 16) & 1u);
  return (u16t)(u >> 16);
}

// 16 bf16 (32B) fragment load (global or LDS pointer)
DEV v16bf ldfrag(const u16t* p) {
  v8u u = *reinterpret_cast<const v8u*>(p);
  return __builtin_bit_cast(v16bf, u);
}

// gfx1250 async global->LDS copy, 16B per lane, tracked by ASYNCcnt
DEV void async_g2l_b128(uint32_t lds_byte_addr, const void* gaddr) {
  unsigned long long ga = (unsigned long long)(uintptr_t)gaddr;
  asm volatile("global_load_async_to_lds_b128 %0, %1, off"
               :: "v"(lds_byte_addr), "v"(ga)
               : "memory");
}

DEV void wait_asynccnt0() {
  asm volatile("s_wait_asynccnt 0" ::: "memory");
}

// ---------------------------------------------------------------------------
// Generic batched WMMA GEMM:  C[batch] = A[batch][M,K] * BT[batch][N,K]^T
//                                        (+ bias[col]) (+ res[row*ldc+col])
// A, BT bf16 with K contiguous (TN form). C f32.
// Block = 256 threads (8 waves), tile 128x64. Wave w owns rows [w*16, w*16+16).
// B tile (64 cols x 32 K) staged into LDS with global_load_async_to_lds_b128.
// K-loop manually unrolled 2x with statically ping-ponged LDS buffers and
// A-fragment registers (K is always a multiple of 64 here), so the async copy
// of step k+1 overlaps the 4 WMMAs of step k with no register rotation.
// C batch offset = (batch/cDiv)*sC0 + (batch%cDiv)*sC1   (head-scatter support)
// ---------------------------------------------------------------------------
constexpr int BROW = 80;         // LDS row pitch in bytes (64B data + 16B bank skew)
constexpr int BBUF = 64 * BROW;  // 5120 B per buffer

DEV void wmma_step(const u16t* sbuf, int lr, int lh, v16bf a, v8f* acc) {
  const char* sbb = (const char*)sbuf + lh * 32;
  v16bf b0 = ldfrag((const u16t*)(sbb + (lr + 0) * BROW));
  v16bf b1 = ldfrag((const u16t*)(sbb + (lr + 16) * BROW));
  v16bf b2 = ldfrag((const u16t*)(sbb + (lr + 32) * BROW));
  v16bf b3 = ldfrag((const u16t*)(sbb + (lr + 48) * BROW));
  acc[0] = __builtin_amdgcn_wmma_f32_16x16x32_bf16(false, a, false, b0, (short)0, acc[0], false, false);
  acc[1] = __builtin_amdgcn_wmma_f32_16x16x32_bf16(false, a, false, b1, (short)0, acc[1], false, false);
  acc[2] = __builtin_amdgcn_wmma_f32_16x16x32_bf16(false, a, false, b2, (short)0, acc[2], false, false);
  acc[3] = __builtin_amdgcn_wmma_f32_16x16x32_bf16(false, a, false, b3, (short)0, acc[3], false, false);
}

__global__ void __launch_bounds__(256)
wmma_gemm_tn(const u16t* __restrict__ A, const u16t* __restrict__ BT,
             const float* __restrict__ bias, const float* __restrict__ res,
             float* __restrict__ Cout,
             int M, int Nn, int K,
             long long sA, long long sBt,
             int cDiv, long long sC0, long long sC1, int ldc,
             int mtiles, int ntiles) {
  __shared__ __align__(16) u16t sB[2 * BBUF / 2];

  const int bpb = mtiles * ntiles;
  const int batch = blockIdx.x / bpb;
  const int rem = blockIdx.x % bpb;
  const int mt = rem / ntiles, nt = rem % ntiles;
  const int lane = threadIdx.x & 31, wave = threadIdx.x >> 5;
  const int m0 = mt * 128 + wave * 16;
  const int n0 = nt * 64;
  const int lr = lane & 15;   // row (A) / col (B) within fragment
  const int lh = lane >> 4;   // K-half selector

  int arow = m0 + lr;
  if (arow > M - 1) arow = M - 1;  // clamp for ragged M
  const u16t* ap = A + (long long)batch * sA + (long long)arow * K + lh * 16;
  const u16t* bg = BT + (long long)batch * sBt + (long long)n0 * K;  // block's B panel

  // staging map: thread t moves 16B; col = t>>2, chunk = t&3
  const int scol = threadIdx.x >> 2, schk = threadIdx.x & 3;
  const u16t* sgp = bg + (long long)scol * K + schk * 8;
  const uint32_t sliceOff = (uint32_t)(scol * BROW + schk * 16);
  const uint32_t lds0 = (uint32_t)(uintptr_t)sB + sliceOff;
  const uint32_t lds1 = lds0 + (uint32_t)BBUF;
  const u16t* sbuf0 = sB;
  const u16t* sbuf1 = sB + BBUF / 2;

  // prologue: stage K-step 0 into buffer 0
  async_g2l_b128(lds0, sgp);
  v16bf a0 = ldfrag(ap);
  v16bf a1;

  v8f acc[4] = {};
  for (int k0 = 0; k0 < K; k0 += 64) {
    // ---- even half: compute buf0[k0], stage buf1[k0+32] ----
    wait_asynccnt0();
    __syncthreads();
    async_g2l_b128(lds1, sgp + k0 + 32);     // k0+32 < K always (K % 64 == 0)
    a1 = ldfrag(ap + k0 + 32);
    __builtin_prefetch(ap + k0 + 320, 0, 1); // -> global_prefetch_b8
    wmma_step(sbuf0, lr, lh, a0, acc);

    // ---- odd half: compute buf1[k0+32], stage buf0[k0+64] ----
    wait_asynccnt0();
    __syncthreads();
    if (k0 + 64 < K) {
      async_g2l_b128(lds0, sgp + k0 + 64);
      a0 = ldfrag(ap + k0 + 64);
    }
    wmma_step(sbuf1, lr, lh, a1, acc);
  }

  float* cb = Cout + (long long)(batch / cDiv) * sC0 + (long long)(batch % cDiv) * sC1;
#pragma unroll
  for (int j = 0; j < 4; ++j) {
    const int col = n0 + j * 16 + lr;
    const float bvv = bias ? bias[col] : 0.0f;
#pragma unroll
    for (int r = 0; r < 8; ++r) {
      const int row = m0 + r + lh * 8;   // C layout: vgpr r, lanes16-31 -> +8
      if (row < M) {
        long long idx = (long long)row * ldc + col;
        float v = acc[j][r] + bvv;
        if (res) v += res[idx];
        cb[idx] = v;
      }
    }
  }
}

// ---------------------------------------------------------------------------
// Elementwise / normalization kernels
// ---------------------------------------------------------------------------
__global__ void transpose_to_bf16(const float* __restrict__ W, u16t* __restrict__ WT,
                                  int K, int Nn) {
  long long i = (long long)blockIdx.x * blockDim.x + threadIdx.x;
  if (i >= (long long)K * Nn) return;
  int n = (int)(i % Nn);
  int k = (int)(i / Nn);
  WT[(long long)n * K + k] = f2bf(W[i]);
}

__global__ void silu_to_bf16(const float* __restrict__ x, u16t* __restrict__ o, int n) {
  int i = blockIdx.x * blockDim.x + threadIdx.x;
  if (i < n) {
    float v = x[i];
    o[i] = f2bf(v / (1.0f + __expf(-v)));
  }
}

// one row per block; LayerNorm then bf16 pack
__global__ void __launch_bounds__(256)
ln_to_bf16(const float* __restrict__ x, const float* __restrict__ g,
           const float* __restrict__ bb, u16t* __restrict__ o, int cols) {
  const long long row = blockIdx.x;
  const float* xr = x + row * cols;
  float s = 0.f, ss = 0.f;
  for (int i = threadIdx.x; i < cols; i += 256) {
    float v = xr[i];
    s += v; ss += v * v;
  }
  __shared__ float red[2][8];
  __shared__ float stats[2];
  const int lane = threadIdx.x & 31, wave = threadIdx.x >> 5;
  for (int m = 16; m >= 1; m >>= 1) { s += __shfl_xor(s, m, 32); ss += __shfl_xor(ss, m, 32); }
  if (lane == 0) { red[0][wave] = s; red[1][wave] = ss; }
  __syncthreads();
  if (threadIdx.x == 0) {
    float S = 0.f, SS = 0.f;
    for (int w = 0; w < 8; ++w) { S += red[0][w]; SS += red[1][w]; }
    float mu = S / cols;
    float var = SS / cols - mu * mu;
    stats[0] = mu; stats[1] = rsqrtf(var + 1e-5f);
  }
  __syncthreads();
  float mu = stats[0], rs = stats[1];
  u16t* orow = o + row * cols;
  for (int i = threadIdx.x; i < cols; i += 256)
    orow[i] = f2bf((xr[i] - mu) * rs * g[i] + bb[i]);
}

// softmax over head channel dim (C=64), repack [B,T,H,C] -> bf16 [B,H,T,C]
__global__ void __launch_bounds__(256)
softmax_q_pack(const float* __restrict__ q, u16t* __restrict__ qbf) {
  using namespace cfg;
  const long long row = blockIdx.x;              // b*T + t
  const int b = (int)(row >> 10), t = (int)(row & 1023);
  const int head = threadIdx.x >> 4, sub = threadIdx.x & 15;  // 16 lanes per head
  const float* p = q + row * D + head * C + sub * 4;
  float v0 = p[0], v1 = p[1], v2 = p[2], v3 = p[3];
  float mx = fmaxf(fmaxf(v0, v1), fmaxf(v2, v3));
  for (int m = 8; m >= 1; m >>= 1) mx = fmaxf(mx, __shfl_xor(mx, m, 16));
  v0 = __expf(v0 - mx); v1 = __expf(v1 - mx);
  v2 = __expf(v2 - mx); v3 = __expf(v3 - mx);
  float sm = v0 + v1 + v2 + v3;
  for (int m = 8; m >= 1; m >>= 1) sm += __shfl_xor(sm, m, 16);
  float inv = 1.0f / sm;
  u16t* orow = qbf + (((long long)b * H + head) * T + t) * C + sub * 4;
  orow[0] = f2bf(v0 * inv); orow[1] = f2bf(v1 * inv);
  orow[2] = f2bf(v2 * inv); orow[3] = f2bf(v3 * inv);
}

// softmax over sequence (N=256) per (b,d); pack transposed bf16 [B, D, N]
__global__ void softmax_k_pack(const float* __restrict__ kf, u16t* __restrict__ kbf) {
  using namespace cfg;
  int gid = blockIdx.x * blockDim.x + threadIdx.x;   // B*D threads
  int b = gid >> 10, d = gid & 1023;
  const float* p = kf + (long long)b * N * D + d;
  float mx = -3.0e38f;
  for (int n = 0; n < N; ++n) mx = fmaxf(mx, p[(long long)n * D]);
  float sm = 0.f;
  for (int n = 0; n < N; ++n) sm += __expf(p[(long long)n * D] - mx);
  float inv = 1.0f / sm;
  u16t* o = kbf + ((long long)b * D + d) * N;
  for (int n = 0; n < N; ++n) o[n] = f2bf(__expf(p[(long long)n * D] - mx) * inv);
}

// v f32 [B,N,D] -> bf16 [B, D, N]  (BT operand for attn = k^T v)
__global__ void pack_v_t(const float* __restrict__ vf, u16t* __restrict__ vbf) {
  using namespace cfg;
  long long i = (long long)blockIdx.x * blockDim.x + threadIdx.x;  // B*N*D
  int d = (int)(i & 1023);
  long long r = i >> 10;
  int n = (int)(r & 255);
  int b = (int)(r >> 8);
  vbf[((long long)b * D + d) * N + n] = f2bf(vf[i]);
}

// attn f32 [BH, C(c), C(d)] -> bf16 transposed [BH, C(d), C(c)]
__global__ void pack_attn_t(const float* __restrict__ attn, u16t* __restrict__ attnT) {
  using namespace cfg;
  long long i = (long long)blockIdx.x * blockDim.x + threadIdx.x;  // BH*C*C
  int dcol = (int)(i & 63);
  long long r = i >> 6;
  int crow = (int)(r & 63);
  long long bh = r >> 6;
  attnT[(bh * C + dcol) * C + crow] = f2bf(attn[i]);
}

// h = LN(y)*(1+scale)+shift ; silu(h) -> bf16 (A operand for out GEMM)
__global__ void __launch_bounds__(256)
film_silu_pack(const float* __restrict__ y, const float* __restrict__ e,
               const float* __restrict__ g, const float* __restrict__ bb,
               u16t* __restrict__ o) {
  using namespace cfg;
  const long long row = blockIdx.x;  // b*T + t
  const int b = (int)(row >> 10);
  const float* yr = y + row * D;
  float s = 0.f, ss = 0.f;
  for (int i = threadIdx.x; i < D; i += 256) {
    float v = yr[i];
    s += v; ss += v * v;
  }
  __shared__ float red[2][8];
  __shared__ float stats[2];
  const int lane = threadIdx.x & 31, wave = threadIdx.x >> 5;
  for (int m = 16; m >= 1; m >>= 1) { s += __shfl_xor(s, m, 32); ss += __shfl_xor(ss, m, 32); }
  if (lane == 0) { red[0][wave] = s; red[1][wave] = ss; }
  __syncthreads();
  if (threadIdx.x == 0) {
    float S = 0.f, SS = 0.f;
    for (int w = 0; w < 8; ++w) { S += red[0][w]; SS += red[1][w]; }
    float mu = S / D;
    float var = SS / D - mu * mu;
    stats[0] = mu; stats[1] = rsqrtf(var + 1e-5f);
  }
  __syncthreads();
  float mu = stats[0], rs = stats[1];
  const float* es = e + (long long)b * D2;
  u16t* orow = o + row * D;
  for (int i = threadIdx.x; i < D; i += 256) {
    float v = (yr[i] - mu) * rs * g[i] + bb[i];
    float h = v * (1.0f + es[i]) + es[D + i];
    orow[i] = f2bf(h / (1.0f + __expf(-h)));
  }
}

// ---------------------------------------------------------------------------
extern "C" void kernel_launch(void* const* d_in, const int* in_sizes, int n_in,
                              void* d_out, int out_size, void* d_ws, size_t ws_size,
                              hipStream_t stream) {
  using namespace cfg;
  const float* x       = (const float*)d_in[0];
  const float* xfp     = (const float*)d_in[1];
  const float* emb     = (const float*)d_in[2];
  const float* norm_g  = (const float*)d_in[3];
  const float* norm_b  = (const float*)d_in[4];
  const float* tnorm_g = (const float*)d_in[5];
  const float* tnorm_b = (const float*)d_in[6];
  const float* Wq      = (const float*)d_in[7];
  const float* bq      = (const float*)d_in[8];
  const float* Wk      = (const float*)d_in[9];
  const float* bk      = (const float*)d_in[10];
  const float* Wv      = (const float*)d_in[11];
  const float* bv      = (const float*)d_in[12];
  const float* emb_W   = (const float*)d_in[13];
  const float* emb_b   = (const float*)d_in[14];
  const float* fnorm_g = (const float*)d_in[15];
  const float* fnorm_b = (const float*)d_in[16];
  const float* out_W   = (const float*)d_in[17];
  const float* out_b   = (const float*)d_in[18];
  float* outp = (float*)d_out;

  char* ws = (char*)d_ws;
  size_t off = 0;
  auto alloc = [&](size_t bytes) -> void* {
    void* p = ws + off;
    off += (bytes + 255) & ~(size_t)255;
    return p;
  };

  // workspace layout (~390 MB with aliasing)
  u16t* WqT   = (u16t*)alloc((size_t)D * D * 2);
  u16t* WkT   = (u16t*)alloc((size_t)TD * D * 2);
  u16t* WvT   = (u16t*)alloc((size_t)TD * D * 2);
  u16t* WeT   = (u16t*)alloc((size_t)TE * D2 * 2);
  u16t* WoT   = (u16t*)alloc((size_t)D * D * 2);
  u16t* xn    = (u16t*)alloc((size_t)B * T * D * 2);
  u16t* xfn   = (u16t*)alloc((size_t)B * N * TD * 2);
  u16t* embs  = (u16t*)alloc((size_t)B * TE * 2);
  float* qf   = (float*)alloc((size_t)B * T * D * 4);
  float* kf   = (float*)alloc((size_t)B * N * D * 4);
  float* vf   = (float*)alloc((size_t)B * N * D * 4);
  u16t* qbf   = (u16t*)alloc((size_t)B * T * D * 2);
  u16t* kbf   = (u16t*)alloc((size_t)B * D * N * 2);
  u16t* vbf   = (u16t*)alloc((size_t)B * D * N * 2);
  float* attnf = (float*)alloc((size_t)B * H * C * C * 4);
  u16t* attnT  = (u16t*)alloc((size_t)B * H * C * C * 2);
  float* ef   = (float*)alloc((size_t)B * D2 * 4);
  float* yf = qf;  // alias: qf dead after softmax_q_pack
  u16t* hs = xn;   // alias: xn dead after q projection GEMM

  const int TPB = 256;
  auto cdiv = [](long long a, long long b) { return (int)((a + b - 1) / b); };
  auto mt128 = [](int M) { return (M + 127) / 128; };

  // weights -> transposed bf16 (TN form); silu(emb) -> bf16
  transpose_to_bf16<<<cdiv((long long)D * D, TPB), TPB, 0, stream>>>(Wq, WqT, D, D);
  transpose_to_bf16<<<cdiv((long long)TD * D, TPB), TPB, 0, stream>>>(Wk, WkT, TD, D);
  transpose_to_bf16<<<cdiv((long long)TD * D, TPB), TPB, 0, stream>>>(Wv, WvT, TD, D);
  transpose_to_bf16<<<cdiv((long long)TE * D2, TPB), TPB, 0, stream>>>(emb_W, WeT, TE, D2);
  transpose_to_bf16<<<cdiv((long long)D * D, TPB), TPB, 0, stream>>>(out_W, WoT, D, D);
  silu_to_bf16<<<cdiv(B * TE, TPB), TPB, 0, stream>>>(emb, embs, B * TE);

  // layernorms -> bf16 activations
  ln_to_bf16<<<B * T, TPB, 0, stream>>>(x, norm_g, norm_b, xn, D);
  ln_to_bf16<<<B * N, TPB, 0, stream>>>(xfp, tnorm_g, tnorm_b, xfn, TD);

  // q/k/v/e projections (WMMA + async-LDS staged B)
  wmma_gemm_tn<<<mt128(B * T) * (D / 64), 256, 0, stream>>>(
      xn, WqT, bq, nullptr, qf, B * T, D, D, 0, 0, 1, 0, 0, D, mt128(B * T), D / 64);
  wmma_gemm_tn<<<mt128(B * N) * (D / 64), 256, 0, stream>>>(
      xfn, WkT, bk, nullptr, kf, B * N, D, TD, 0, 0, 1, 0, 0, D, mt128(B * N), D / 64);
  wmma_gemm_tn<<<mt128(B * N) * (D / 64), 256, 0, stream>>>(
      xfn, WvT, bv, nullptr, vf, B * N, D, TD, 0, 0, 1, 0, 0, D, mt128(B * N), D / 64);
  wmma_gemm_tn<<<(D2 / 64), 256, 0, stream>>>(
      embs, WeT, emb_b, nullptr, ef, B, D2, TE, 0, 0, 1, 0, 0, D2, 1, D2 / 64);

  // dual softmax + operand packing
  softmax_q_pack<<<B * T, 256, 0, stream>>>(qf, qbf);
  softmax_k_pack<<<(B * D) / 256, 256, 0, stream>>>(kf, kbf);
  pack_v_t<<<cdiv((long long)B * N * D, TPB), TPB, 0, stream>>>(vf, vbf);

  // attn = k^T v : 512 batched 64x64x256 (WMMA)
  wmma_gemm_tn<<<B * H, 256, 0, stream>>>(
      kbf, vbf, nullptr, nullptr, attnf, C, C, N,
      (long long)C * N, (long long)C * N, 1, (long long)C * C, 0, C, 1, 1);
  pack_attn_t<<<cdiv((long long)B * H * C * C, TPB), TPB, 0, stream>>>(attnf, attnT);

  // y = q @ attn : 512 batched 1024x64x64, scattered into [B,T,D] (WMMA)
  wmma_gemm_tn<<<B * H * mt128(T), 256, 0, stream>>>(
      qbf, attnT, nullptr, nullptr, yf, T, C, C,
      (long long)T * C, (long long)C * C, H, (long long)T * D, (long long)C, D, mt128(T), 1);

  // FiLM + silu -> bf16
  film_silu_pack<<<B * T, 256, 0, stream>>>(yf, ef, fnorm_g, fnorm_b, hs);

  // out = x + silu(h) @ out_W + out_b (WMMA, fused residual)
  wmma_gemm_tn<<<mt128(B * T) * (D / 64), 256, 0, stream>>>(
      hs, WoT, out_b, x, outp, B * T, D, D, 0, 0, 1, 0, 0, D, mt128(B * T), D / 64);
}
